// OneStageOVDet_61160334295433
// MI455X (gfx1250) — matrix operand order repaired and checked by reading, not compiled, not run on hardware
//
#include <hip/hip_runtime.h>
#include <hip/hip_bf16.h>

typedef __attribute__((ext_vector_type(16))) _Float16 v16h;
typedef __attribute__((ext_vector_type(8)))  _Float16 v8h;
typedef __attribute__((ext_vector_type(8)))  float    v8f;

// Problem constants
#define CB   32
#define CT   128
#define CD   512
#define CNW  2000
#define CNC  200
#define CS   8
#define CL   4
#define CH   8
#define CFF  2048
#define CHD  64
#define NWC  250   // cross-attention NW chunk

enum { GF_BIAS = 1, GF_ACCUM = 2, GF_RELU = 4 };

struct GemmP {
  const float* A;
  const float* Bm;
  const float* bias;
  float*       C;
  int lda, ldbk, ldbn, ldc;
  int M, N, K;
  int bdiv;
  long long sA1, sA2, sB1, sB2, sC1, sC2;
  int rows_per_b, shift, flags;
  float alpha;
};

__device__ __forceinline__ v16h cat8(v8h x, v8h y) {
  return __builtin_shufflevector(x, y, 0,1,2,3,4,5,6,7,8,9,10,11,12,13,14,15);
}

__device__ __forceinline__ v8h cvt8(float4 f0, float4 f1) {
  v8h h;
  h[0] = (_Float16)f0.x; h[1] = (_Float16)f0.y; h[2] = (_Float16)f0.z; h[3] = (_Float16)f0.w;
  h[4] = (_Float16)f1.x; h[5] = (_Float16)f1.y; h[6] = (_Float16)f1.z; h[7] = (_Float16)f1.w;
  return h;
}

// ---------------------------------------------------------------------------
// Generic batched WMMA GEMM: C[m,n] (+)= alpha * sum_k A[m,k]*B[k,n] (+bias)(relu)
// Block: 256 threads (8 wave32), tile 128(M) x 64(N), K-step 32.
// Each wave: 2x2 grid of v_wmma_f32_16x16x32_f16, f32 accumulation.
// Edge handling is branchless: addresses are clamped into valid range and the
// loaded value is zeroed with a multiplicative mask (no per-element predicated
// loads -> loads stay pipelined even on edge tiles).
// NOTE: all A operands have lda%4==0 and 16B-aligned bases; all ldbk==1 B
// operands have ldbn%4==0 — required by the float4 fast paths below.
// ---------------------------------------------------------------------------
__launch_bounds__(256)
__global__ void wmma_gemm_kernel(GemmP p) {
  const int z  = blockIdx.z;
  const int zq = z / p.bdiv;
  const int zr = z % p.bdiv;
  const float* __restrict__ Ab = p.A  + zq * p.sA1 + zr * p.sA2;
  const float* __restrict__ Bb = p.Bm + zq * p.sB1 + zr * p.sB2;
  float*       Cb = p.C  + zq * p.sC1 + zr * p.sC2;

  __shared__ alignas(16) _Float16 As[128][40];  // padded: 80B rows, 16B aligned
  __shared__ alignas(16) _Float16 Bs[64][40];

  const int tid     = threadIdx.x;
  const int m_block = blockIdx.y * 128;
  const int n_block = blockIdx.x * 64;

  const int wave = tid >> 5;
  const int lane = tid & 31;
  const int wm   = wave & 3;   // 4 waves along M
  const int wn   = wave >> 2;  // 2 waves along N
  const int lo   = lane & 15;
  const int hi   = lane >> 4;

  v8f acc00 = {}, acc01 = {}, acc10 = {}, acc11 = {};

  // A-tile loader: thread -> (row = tid/2, 16 cols starting at (tid&1)*16)
  const int a_r  = tid >> 1;
  const int a_c0 = (tid & 1) << 4;
  const int gr   = m_block + a_r;
  bool a_ok;
  long long a_off;          // always a VALID row offset (clamped); a_ok gates value
  if (p.rows_per_b > 0) {   // conv temporal shift: row -> (b, t); src t = t+shift
    int gr_c = gr < p.M ? gr : p.M - 1;
    int b_ = gr_c / p.rows_per_b;
    int t_ = gr_c - b_ * p.rows_per_b;
    int ts = t_ + p.shift;
    a_ok  = (gr < p.M) && (ts >= 0) && (ts < p.rows_per_b);
    int ts_c = ts < 0 ? 0 : (ts >= p.rows_per_b ? p.rows_per_b - 1 : ts);
    a_off = (long long)(b_ * p.rows_per_b + ts_c) * (long long)p.lda;
  } else {
    a_ok  = (gr < p.M);
    a_off = (long long)(gr < p.M ? gr : p.M - 1) * (long long)p.lda;
  }
  const float amask = a_ok ? 1.0f : 0.0f;

  // B-tile loader: thread -> (col n = tid/4, 8 k's starting at (tid&3)*8)
  const int b_n  = tid >> 2;
  const int b_k0 = (tid & 3) << 3;
  const int gn   = n_block + b_n;
  const bool b_ok = (gn < p.N);
  const float bmask = b_ok ? 1.0f : 0.0f;
  const long long b_coloff = (long long)(gn < p.N ? gn : p.N - 1) * (long long)p.ldbn;

  for (int k0 = 0; k0 < p.K; k0 += 32) {
    // ---- A tile -> LDS (f32 -> f16) ----
    {
      const float* aptr = Ab + a_off + (k0 + a_c0);
      if (a_ok && (k0 + a_c0 + 16 <= p.K)) {
        // fast path: 4x b128 loads, 2x 16B LDS stores
        const float4* ap4 = (const float4*)aptr;
        float4 f0 = ap4[0], f1 = ap4[1], f2 = ap4[2], f3 = ap4[3];
        *(v8h*)&As[a_r][a_c0]     = cvt8(f0, f1);
        *(v8h*)&As[a_r][a_c0 + 8] = cvt8(f2, f3);
      } else {
        // branchless edge path: clamped address + multiplicative mask
        const float* arow = Ab + a_off;
        v8h h0, h1v;
#pragma unroll
        for (int j = 0; j < 8; ++j) {
          int kk = k0 + a_c0 + j;
          int kc = kk < p.K ? kk : p.K - 1;
          float m = (kk < p.K) ? amask : 0.0f;
          h0[j] = (_Float16)(arow[kc] * m);
        }
#pragma unroll
        for (int j = 0; j < 8; ++j) {
          int kk = k0 + a_c0 + 8 + j;
          int kc = kk < p.K ? kk : p.K - 1;
          float m = (kk < p.K) ? amask : 0.0f;
          h1v[j] = (_Float16)(arow[kc] * m);
        }
        *(v8h*)&As[a_r][a_c0]     = h0;
        *(v8h*)&As[a_r][a_c0 + 8] = h1v;
      }
    }
    // ---- B tile -> LDS (f32 -> f16) ----
    {
      const long long bbase = (long long)(k0 + b_k0) * (long long)p.ldbk + b_coloff;
      const bool fullk = (k0 + b_k0 + 8 <= p.K);
      v8h h;
      if (b_ok && fullk && p.ldbk == 1) {
        // contiguous-K fast path: 2x b128 loads
        const float4* bp4 = (const float4*)(Bb + bbase);
        float4 f0 = bp4[0], f1 = bp4[1];
        h = cvt8(f0, f1);
      } else if (b_ok && fullk) {
        // strided but fully in range: 8 independent unconditional loads
        float v0 = Bb[bbase + 0LL * p.ldbk];
        float v1 = Bb[bbase + 1LL * p.ldbk];
        float v2 = Bb[bbase + 2LL * p.ldbk];
        float v3 = Bb[bbase + 3LL * p.ldbk];
        float v4 = Bb[bbase + 4LL * p.ldbk];
        float v5 = Bb[bbase + 5LL * p.ldbk];
        float v6 = Bb[bbase + 6LL * p.ldbk];
        float v7 = Bb[bbase + 7LL * p.ldbk];
        h[0] = (_Float16)v0; h[1] = (_Float16)v1; h[2] = (_Float16)v2; h[3] = (_Float16)v3;
        h[4] = (_Float16)v4; h[5] = (_Float16)v5; h[6] = (_Float16)v6; h[7] = (_Float16)v7;
      } else {
        // branchless edge path: clamped address + multiplicative mask
#pragma unroll
        for (int j = 0; j < 8; ++j) {
          int kk = k0 + b_k0 + j;
          int kc = kk < p.K ? kk : p.K - 1;
          float m = (kk < p.K) ? bmask : 0.0f;
          h[j] = (_Float16)(Bb[(long long)kc * (long long)p.ldbk + b_coloff] * m);
        }
      }
      *(v8h*)&Bs[b_n][b_k0] = h;
    }
    __syncthreads();

    // Fragments per ISA 7.12.2:
    // A (16x32 f16): lane=M row; elems 0..7 -> K = 8*hi + e ; elems 8..15 -> K = 16 + 8*hi + (e-8)
    // B (32x16 f16): lane=N col; elems 0..7 -> K = 8*hi + e ; elems 8..15 -> K = 16 + 8*hi + (e-8)
    v16h a0, a1, b0, b1;
    {
      const int m0 = wm * 32 + lo;
      v8h l0 = *(const v8h*)&As[m0][hi * 8];
      v8h h0 = *(const v8h*)&As[m0][16 + hi * 8];
      a0 = cat8(l0, h0);
      const int m1 = m0 + 16;
      v8h l1 = *(const v8h*)&As[m1][hi * 8];
      v8h h1 = *(const v8h*)&As[m1][16 + hi * 8];
      a1 = cat8(l1, h1);
      const int n0 = wn * 32 + lo;
      v8h c0 = *(const v8h*)&Bs[n0][hi * 8];
      v8h c1 = *(const v8h*)&Bs[n0][16 + hi * 8];
      b0 = cat8(c0, c1);
      const int n1 = n0 + 16;
      v8h d0 = *(const v8h*)&Bs[n1][hi * 8];
      v8h d1 = *(const v8h*)&Bs[n1][16 + hi * 8];
      b1 = cat8(d0, d1);
    }
    acc00 = __builtin_amdgcn_wmma_f32_16x16x32_f16(false, a0, false, b0, (short)0, acc00, false, false);
    acc01 = __builtin_amdgcn_wmma_f32_16x16x32_f16(false, a0, false, b1, (short)0, acc01, false, false);
    acc10 = __builtin_amdgcn_wmma_f32_16x16x32_f16(false, a1, false, b0, (short)0, acc10, false, false);
    acc11 = __builtin_amdgcn_wmma_f32_16x16x32_f16(false, a1, false, b1, (short)0, acc11, false, false);
    __syncthreads();
  }

  // Epilogue. C/D layout: VGPR r, lane-half hi -> M = r + 8*hi ; N = lane&15.
  const bool full_m = (m_block + 128 <= p.M);
  union { v8f v; float f[8]; } u;
#pragma unroll
  for (int ij = 0; ij < 4; ++ij) {
    const int i = ij >> 1, j = ij & 1;
    u.v = (ij == 0) ? acc00 : (ij == 1) ? acc01 : (ij == 2) ? acc10 : acc11;
    const int gm_base = m_block + wm * 32 + i * 16 + hi * 8;
    const int gn_     = n_block + wn * 32 + j * 16 + lo;
    if (gn_ < p.N) {
      const float bvv = (p.flags & GF_BIAS) ? p.bias[gn_] : 0.0f;
      if (full_m) {
        float* cp = Cb + (long long)gm_base * (long long)p.ldc + gn_;
#pragma unroll
        for (int r = 0; r < 8; ++r) {
          float o = u.f[r] * p.alpha + bvv;
          if (p.flags & GF_ACCUM) o += cp[(long long)r * p.ldc];
          if (p.flags & GF_RELU)  o = fmaxf(o, 0.0f);
          cp[(long long)r * p.ldc] = o;
        }
      } else {
#pragma unroll
        for (int r = 0; r < 8; ++r) {
          const int gm = gm_base + r;
          if (gm < p.M) {
            const long long ci = (long long)gm * (long long)p.ldc + gn_;
            float o = u.f[r] * p.alpha + bvv;
            if (p.flags & GF_ACCUM) o += Cb[ci];
            if (p.flags & GF_RELU)  o = fmaxf(o, 0.0f);
            Cb[ci] = o;
          }
        }
      }
    }
  }
}

// ---------------------------------------------------------------------------
// Elementwise add with optional broadcast of second operand (index % bmod)
// ---------------------------------------------------------------------------
__global__ void add_bcast_kernel(const float* a, const float* b, float* c,
                                 long long n, long long bmod) {
  long long i = (long long)blockIdx.x * blockDim.x + threadIdx.x;
  if (i < n) c[i] = a[i] + b[bmod ? (i % bmod) : i];
}

// ---------------------------------------------------------------------------
// out = LayerNorm(x + y; g, b)   (safe when out == x)
// ---------------------------------------------------------------------------
__global__ void add_ln_kernel(const float* x, const float* y,
                              const float* g, const float* b,
                              float* out, int Dv) {
  __shared__ float s1[256], s2[256];
  const long long base = (long long)blockIdx.x * Dv;
  float sum = 0.f, sq = 0.f;
  for (int d = threadIdx.x; d < Dv; d += 256) {
    float v = x[base + d] + y[base + d];
    sum += v; sq += v * v;
  }
  s1[threadIdx.x] = sum; s2[threadIdx.x] = sq;
  __syncthreads();
  for (int s = 128; s > 0; s >>= 1) {
    if ((int)threadIdx.x < s) { s1[threadIdx.x] += s1[threadIdx.x + s]; s2[threadIdx.x] += s2[threadIdx.x + s]; }
    __syncthreads();
  }
  const float mean = s1[0] / Dv;
  const float var  = s2[0] / Dv - mean * mean;
  const float inv  = rsqrtf(var + 1e-5f);
  for (int d = threadIdx.x; d < Dv; d += 256) {
    float v = x[base + d] + y[base + d];
    out[base + d] = (v - mean) * inv * g[d] + b[d];
  }
}

// ---------------------------------------------------------------------------
// Row softmax, row length Ln <= 128 (one block of 128 threads per row)
// ---------------------------------------------------------------------------
__global__ void softmax128_kernel(float* s, int Ln) {
  __shared__ float red[128];
  const long long base = (long long)blockIdx.x * Ln;
  const int t = threadIdx.x;
  float v = (t < Ln) ? s[base + t] : -3.4e38f;
  red[t] = v; __syncthreads();
  for (int k = 64; k > 0; k >>= 1) { if (t < k) red[t] = fmaxf(red[t], red[t + k]); __syncthreads(); }
  const float mx = red[0]; __syncthreads();
  float e = (t < Ln) ? __expf(v - mx) : 0.f;
  red[t] = e; __syncthreads();
  for (int k = 64; k > 0; k >>= 1) { if (t < k) red[t] += red[t + k]; __syncthreads(); }
  const float inv = 1.0f / red[0];
  if (t < Ln) s[base + t] = e * inv;
}

// ---------------------------------------------------------------------------
// conv5 (128 -> 1 channels, k=3, pad 1) + sigmoid -> frames_cls (B*T)
// ---------------------------------------------------------------------------
__global__ void conv5_sig_kernel(const float* h4, const float* w, const float* b5,
                                 float* out, int Bv, int Tv, int Ci) {
  int idx = blockIdx.x * blockDim.x + threadIdx.x;
  if (idx >= Bv * Tv) return;
  int b_ = idx / Tv, t = idx - b_ * Tv;
  float acc = b5[0];
  for (int dk = 0; dk < 3; ++dk) {
    int ts = t + dk - 1;
    if (ts < 0 || ts >= Tv) continue;
    const float* hp = h4 + (long long)(b_ * Tv + ts) * Ci;
    for (int i = 0; i < Ci; ++i) acc += w[i * 3 + dk] * hp[i];
  }
  out[idx] = 1.0f / (1.0f + __expf(-acc));
}

// ---------------------------------------------------------------------------
// Segment softmax-pooling: seg_feats[b,s,:] = sum_t softmax(masked w)[t]*vid[b,t,:]
// One 128-thread block per (b,s)
// ---------------------------------------------------------------------------
__global__ void seg_pool_kernel(const float* weights, const int* steds,
                                const float* vid, float* seg_feats,
                                int Tv, int Dv, int Sv) {
  __shared__ float w_sh[128];
  __shared__ float red[128];
  const int bs = blockIdx.x;
  const int b_ = bs / Sv;
  const int st = steds[bs * 2 + 0];
  const int ed = steds[bs * 2 + 1];
  const int t  = threadIdx.x;
  const bool in = (t >= st) && (t <= ed);
  float v = in ? weights[b_ * Tv + t] : -3.4e38f;
  red[t] = v; __syncthreads();
  for (int k = 64; k > 0; k >>= 1) { if (t < k) red[t] = fmaxf(red[t], red[t + k]); __syncthreads(); }
  const float mx = red[0]; __syncthreads();
  float e = in ? __expf(v - mx) : 0.f;
  red[t] = e; __syncthreads();
  for (int k = 64; k > 0; k >>= 1) { if (t < k) red[t] += red[t + k]; __syncthreads(); }
  w_sh[t] = e / red[0];
  __syncthreads();
  for (int d = threadIdx.x; d < Dv; d += 128) {
    float acc = 0.f;
    for (int tt = st; tt <= ed && tt < Tv; ++tt)
      acc += w_sh[tt] * vid[((long long)b_ * Tv + tt) * Dv + d];
    seg_feats[(long long)bs * Dv + d] = acc;
  }
}

// ---------------------------------------------------------------------------
// Host-side launch helper
// ---------------------------------------------------------------------------
static void gemm(hipStream_t st,
                 const float* A, int lda,
                 const float* Bm, int ldbk, int ldbn,
                 const float* bias, float* C, int ldc,
                 int M, int N, int K, int flags, float alpha,
                 int batch = 1, int bdiv = 1,
                 long long sA1 = 0, long long sA2 = 0,
                 long long sB1 = 0, long long sB2 = 0,
                 long long sC1 = 0, long long sC2 = 0,
                 int rows_per_b = 0, int shift = 0) {
  GemmP p;
  p.A = A; p.Bm = Bm; p.bias = bias; p.C = C;
  p.lda = lda; p.ldbk = ldbk; p.ldbn = ldbn; p.ldc = ldc;
  p.M = M; p.N = N; p.K = K; p.bdiv = bdiv;
  p.sA1 = sA1; p.sA2 = sA2; p.sB1 = sB1; p.sB2 = sB2; p.sC1 = sC1; p.sC2 = sC2;
  p.rows_per_b = rows_per_b; p.shift = shift; p.flags = flags; p.alpha = alpha;
  dim3 g((N + 63) / 64, (M + 127) / 128, batch);
  wmma_gemm_kernel<<<g, dim3(256), 0, st>>>(p);
}

extern "C" void kernel_launch(void* const* d_in, const int* in_sizes, int n_in,
                              void* d_out, int out_size, void* d_ws, size_t ws_size,
                              hipStream_t stream) {
  const float* vid   = (const float*)d_in[0];
  // d_in[1] segs_feats: unused by reference
  const float* actnF = (const float*)d_in[2];
  const float* actnC = (const float*)d_in[3];
  const float* pos   = (const float*)d_in[4];
  const float* Wq = (const float*)d_in[5];  const float* bq = (const float*)d_in[6];
  const float* Wk = (const float*)d_in[7];  const float* bk = (const float*)d_in[8];
  const float* Wv = (const float*)d_in[9];  const float* bv = (const float*)d_in[10];
  const float* Wo = (const float*)d_in[11]; const float* bo = (const float*)d_in[12];
  const float* W1 = (const float*)d_in[13]; const float* b1 = (const float*)d_in[14];
  const float* W2 = (const float*)d_in[15]; const float* b2 = (const float*)d_in[16];
  const float* ln1g = (const float*)d_in[17]; const float* ln1b = (const float*)d_in[18];
  const float* ln2g = (const float*)d_in[19]; const float* ln2b = (const float*)d_in[20];
  const float* pqW = (const float*)d_in[21]; const float* pqb = (const float*)d_in[22];
  const float* pkW = (const float*)d_in[23]; const float* pkb = (const float*)d_in[24];
  const float* pvW = (const float*)d_in[25]; const float* pvb = (const float*)d_in[26];
  const float* c1w = (const float*)d_in[27]; const float* c1b = (const float*)d_in[28];
  const float* c2w = (const float*)d_in[29]; const float* c2b = (const float*)d_in[30];
  const float* c3w = (const float*)d_in[31]; const float* c3b = (const float*)d_in[32];
  const float* c4w = (const float*)d_in[33]; const float* c4b = (const float*)d_in[34];
  const float* c5w = (const float*)d_in[35]; const float* c5b = (const float*)d_in[36];
  const int*   steds = (const int*)d_in[37];

  float* out = (float*)d_out;
  float* out_frames = out;                          // B*1*T = 4096
  float* out_logits = out + (long long)CB * CT;     // B*S*NC = 51200
  float* out_fp     = out_logits + (long long)CB * CS * CNC; // B*T*NW

  const long long NTD = (long long)CB * CT * CD;    // 2,097,152

  float* ws  = (float*)d_ws;
  float* x   = ws;                  // persists through transformer
  float* vfp = ws + NTD;            // persists through cross-attn
  float* scr = ws + 2 * NTD;        // phase-aliased scratch region

  // Phase 1 (transformer) layout in scr
  float* qb  = scr;
  float* kb  = qb + NTD;
  float* vb  = kb + NTD;
  float* sb  = vb + NTD;                               // B*H*T*T
  float* ob  = sb + (long long)CB * CH * CT * CT;
  float* mid = ob + NTD;                               // B*T*FF
  float* yb  = mid + (long long)CB * CT * CFF;

  // Phase 2 (cross-attn), aliases scr
  float* Qc   = scr;                                   // NW*D
  float* Kc   = Qc + (long long)CNW * CD;
  float* Vc   = Kc + NTD;
  float* csb  = Vc + NTD;                              // B*H*NWC*T
  float* updc = csb + (long long)CB * CH * NWC * CT;   // B*NWC*D

  // Phase 3 (convs), aliases scr
  float* h1   = scr;                                   // B*T*1024
  float* h2   = h1 + (long long)CB * CT * 1024;
  float* h3   = h2 + (long long)CB * CT * 512;
  float* h4   = h3 + (long long)CB * CT * 256;
  float* segf = h4 + (long long)CB * CT * 128;         // B*S*D

  const int BT = CB * CT;

  // x = vid + pos (pos broadcast over batch)
  add_bcast_kernel<<<(unsigned)((NTD + 255) / 256), 256, 0, stream>>>(vid, pos, x, NTD, (long long)CT * CD);

  // ------------------ Transformer encoder ------------------
  for (int l = 0; l < CL; ++l) {
    const long long wOff = (long long)l * CD * CD;
    gemm(stream, x, CD, Wq + wOff, CD, 1, bq + l * CD, qb, CD, BT, CD, CD, GF_BIAS, 1.f);
    gemm(stream, x, CD, Wk + wOff, CD, 1, bk + l * CD, kb, CD, BT, CD, CD, GF_BIAS, 1.f);
    gemm(stream, x, CD, Wv + wOff, CD, 1, bv + l * CD, vb, CD, BT, CD, CD, GF_BIAS, 1.f);
    // scores[b,h,q,k] = (q . k) / sqrt(HD)
    gemm(stream, qb, CD, kb, 1, CD, nullptr, sb, CT, CT, CT, CHD, 0, 0.125f,
         CB * CH, CH,
         (long long)CT * CD, CHD,
         (long long)CT * CD, CHD,
         (long long)CH * CT * CT, (long long)CT * CT);
    softmax128_kernel<<<CB * CH * CT, 128, 0, stream>>>(sb, CT);
    // o[b,q,h,:] = a @ v
    gemm(stream, sb, CT, vb, CD, 1, nullptr, ob, CD, CT, CHD, CT, 0, 1.f,
         CB * CH, CH,
         (long long)CH * CT * CT, (long long)CT * CT,
         (long long)CT * CD, CHD,
         (long long)CT * CD, CHD);
    gemm(stream, ob, CD, Wo + wOff, CD, 1, bo + l * CD, yb, CD, BT, CD, CD, GF_BIAS, 1.f);
    add_ln_kernel<<<BT, 256, 0, stream>>>(x, yb, ln1g + l * CD, ln1b + l * CD, x, CD);
    gemm(stream, x, CD, W1 + (long long)l * CD * CFF, CFF, 1, b1 + l * CFF,
         mid, CFF, BT, CFF, CD, GF_BIAS | GF_RELU, 1.f);
    gemm(stream, mid, CFF, W2 + (long long)l * CFF * CD, CD, 1, b2 + l * CD,
         yb, CD, BT, CD, CFF, GF_BIAS, 1.f);
    add_ln_kernel<<<BT, 256, 0, stream>>>(x, yb, ln2g + l * CD, ln2b + l * CD, x, CD);
  }

  // vfp = vid + x
  add_bcast_kernel<<<(unsigned)((NTD + 255) / 256), 256, 0, stream>>>(vid, x, vfp, NTD, 0);

  // ------------------ Cross-attention (chunked over NW) ------------------
  gemm(stream, actnC, CD, pqW, CD, 1, pqb, Qc, CD, CNW, CD, CD, GF_BIAS, 1.f);
  gemm(stream, vfp,   CD, pkW, CD, 1, pkb, Kc, CD, BT,  CD, CD, GF_BIAS, 1.f);
  gemm(stream, vfp,   CD, pvW, CD, 1, pvb, Vc, CD, BT,  CD, CD, GF_BIAS, 1.f);

  for (int nc = 0; nc < CNW; nc += NWC) {
    const int Mn = NWC;
    // s[b,h,n,t] = (Q[n,h,:] . K[b,t,h,:]) / 8
    gemm(stream, Qc + (long long)nc * CD, CD, Kc, 1, CD, nullptr, csb, CT,
         Mn, CT, CHD, 0, 0.125f,
         CB * CH, CH,
         0LL, (long long)CHD,
         (long long)CT * CD, (long long)CHD,
         (long long)CH * Mn * CT, (long long)Mn * CT);
    softmax128_kernel<<<CB * CH * Mn, 128, 0, stream>>>(csb, CT);
    // upd[b,n,h,:] = a @ V
    gemm(stream, csb, CT, Vc, CD, 1, nullptr, updc, CD,
         Mn, CHD, CT, 0, 1.f,
         CB * CH, CH,
         (long long)CH * Mn * CT, (long long)Mn * CT,
         (long long)CT * CD, (long long)CHD,
         (long long)Mn * CD, (long long)CHD);
    // feats_proj[b,t,nc+n] = vfp[b,t,:] . upd[b,n,:]
    gemm(stream, vfp, CD, updc, 1, CD, nullptr, out_fp + nc, CNW,
         CT, Mn, CD, 0, 1.f,
         CB, 1,
         (long long)CT * CD, 0,
         (long long)Mn * CD, 0,
         (long long)CT * CNW, 0);
  }

  // ------------------ Conv stack (each conv = 3 shifted GEMM passes) ------------------
  struct ConvDesc { const float* in; int Cin; const float* w; const float* b; float* out; int Cout; };
  ConvDesc convs[4] = {
    { out_fp, CNW,  c1w, c1b, h1, 1024 },
    { h1,     1024, c2w, c2b, h2, 512  },
    { h2,     512,  c3w, c3b, h3, 256  },
    { h3,     256,  c4w, c4b, h4, 128  },
  };
  for (int c = 0; c < 4; ++c) {
    for (int dk = 0; dk < 3; ++dk) {
      int flags = (dk == 0 ? GF_BIAS : GF_ACCUM) | (dk == 2 ? GF_RELU : 0);
      gemm(stream, convs[c].in, convs[c].Cin,
           convs[c].w + dk, 3, convs[c].Cin * 3,
           convs[c].b, convs[c].out, convs[c].Cout,
           BT, convs[c].Cout, convs[c].Cin, flags, 1.f,
           1, 1, 0, 0, 0, 0, 0, 0,
           /*rows_per_b=*/CT, /*shift=*/dk - 1);
    }
  }
  conv5_sig_kernel<<<(BT + 127) / 128, 128, 0, stream>>>(h4, c5w, c5b, out_frames, CB, CT, 128);

  // ------------------ Segment pooling + action logits ------------------
  seg_pool_kernel<<<CB * CS, 128, 0, stream>>>(out_frames, steds, vid, segf, CT, CD, CS);
  gemm(stream, segf, CD, actnF, 1, CD, nullptr, out_logits, CNC,
       CB * CS, CNC, CD, 0, 1.f);
}
